// DecoderRNN_42683384987766
// MI455X (gfx1250) — matrix operand order repaired and checked by reading, not compiled, not run on hardware
//
#include <hip/hip_runtime.h>
#include <math.h>

// ---------------- problem constants ----------------
static constexpr int Bc = 512;
static constexpr int Sc = 64;
static constexpr int Hc = 1024;
static constexpr int Vc = 4096;
static constexpr int Tc = 6;
static constexpr int SOS = 1;

// d_out layout (flat, concatenated in return order)
static constexpr size_t OUT_LOGITS = 0;                               // B*T*V
static constexpr size_t OUT_HT     = (size_t)Bc * Tc * Vc;            // B*H
static constexpr size_t OUT_CT     = OUT_HT + (size_t)Bc * Hc;        // B*H
static constexpr size_t OUT_ATTN   = OUT_CT + (size_t)Bc * Hc;        // B*T*S

// ---------------- types ----------------
typedef __attribute__((ext_vector_type(16))) __bf16          v16bf;
typedef __attribute__((ext_vector_type(8)))  float           v8f;
typedef __attribute__((ext_vector_type(8)))  unsigned short  us8;
typedef __attribute__((ext_vector_type(16))) unsigned short  us16;

union ABpack { us8 h[2]; us16 w; v16bf v; };

__device__ inline unsigned short f32_to_bf16(float f) {
  unsigned int u = __float_as_uint(f);
  u = u + 0x7FFFu + ((u >> 16) & 1u);   // round-to-nearest-even
  return (unsigned short)(u >> 16);
}

__device__ inline float sigmoidf_(float x) { return 1.0f / (1.0f + __expf(-x)); }

__device__ inline float block_reduce_sum(float v, float* sh) {
  int tid = threadIdx.x;
  sh[tid] = v; __syncthreads();
  for (int off = blockDim.x >> 1; off > 0; off >>= 1) {
    if (tid < off) sh[tid] += sh[tid + off];
    __syncthreads();
  }
  float r = sh[0]; __syncthreads();
  return r;
}

// ---------------- elementwise / prep kernels ----------------
__global__ void k_f32_to_bf16(const float* __restrict__ src,
                              unsigned short* __restrict__ dst, int n) {
  int i = blockIdx.x * blockDim.x + threadIdx.x;
  if (i < n) dst[i] = f32_to_bf16(src[i]);
}

// W_cat[n, 0:2H] = w_ih[n,:], W_cat[n, 2H:3H] = w_hh[n,:]  (bf16)
__global__ void k_build_wcat(const float* __restrict__ w_ih,
                             const float* __restrict__ w_hh,
                             unsigned short* __restrict__ wcat) {
  int i = blockIdx.x * blockDim.x + threadIdx.x;
  const int KC = 3 * Hc;
  if (i >= 4 * Hc * KC) return;
  int n = i / KC, k = i % KC;
  float v = (k < 2 * Hc) ? w_ih[(size_t)n * 2 * Hc + k]
                         : w_hh[(size_t)n * Hc + (k - 2 * Hc)];
  wcat[i] = f32_to_bf16(v);
}

__global__ void k_bias_cat(const float* __restrict__ b_ih,
                           const float* __restrict__ b_hh,
                           float* __restrict__ out) {
  int i = blockIdx.x * blockDim.x + threadIdx.x;
  if (i < 4 * Hc) out[i] = b_ih[i] + b_hh[i];
}

// ---------------- bf16 WMMA GEMM: C(M,N) = A(M,K) @ W(N,K)^T + bias ----------------
// block: 256 thr = 8 waves as 4(M) x 2(N); wave tile 32x64 (2x4 WMMA tiles);
// block tile 128x128. 8 WMMAs per K-step vs 12 b128-equivalent loads (~21 FLOP/B).
__global__ __launch_bounds__(256)
void k_gemm_bf16(const unsigned short* __restrict__ A, int lda,
                 const unsigned short* __restrict__ W, int ldw,
                 const float* __restrict__ bias,
                 float* __restrict__ C, int ldc, int K) {
  const int lane = threadIdx.x & 31;
  const int wave = threadIdx.x >> 5;
  const int wy = wave >> 1, wx = wave & 1;
  const int m0 = blockIdx.y * 128 + wy * 32;
  const int n0 = blockIdx.x * 128 + wx * 64;
  const int lrow = lane & 15;
  const int lhi  = lane >> 4;

  v8f acc[2][4] = {};

  // A layout (16-bit 16x32): lanes 0-15 -> K 0..15, lanes 16-31 -> K 16..31 (contiguous)
  const unsigned short* Ab = A + (size_t)(m0 + lrow) * lda + lhi * 16;
  // B layout (16-bit 32x16): lanes 0-15 -> K 0..15, lanes 16-31 -> K 16..31 (contiguous)
  const unsigned short* Wb = W + (size_t)(n0 + lrow) * ldw + lhi * 16;

  for (int k0 = 0; k0 < K; k0 += 32) {
    ABpack a[2], b[4];
#pragma unroll
    for (int i = 0; i < 2; ++i) {
      const unsigned short* ar = Ab + (size_t)(16 * i) * lda + k0;
      a[i].h[0] = *(const us8*)(ar);
      a[i].h[1] = *(const us8*)(ar + 8);
    }
#pragma unroll
    for (int j = 0; j < 4; ++j)
      b[j].w = *(const us16*)(Wb + (size_t)(16 * j) * ldw + k0);
    if (k0 + 32 < K) {
      __builtin_prefetch(Ab + k0 + 32, 0, 3);   // near-cache prefetch of next K tile
      __builtin_prefetch(Wb + k0 + 32, 0, 3);
    }
#pragma unroll
    for (int i = 0; i < 2; ++i)
#pragma unroll
      for (int j = 0; j < 4; ++j)
        acc[i][j] = __builtin_amdgcn_wmma_f32_16x16x32_bf16(
            false, a[i].v, false, b[j].v, (short)0, acc[i][j], false, false);
  }

  // C/D layout: VGPR v -> row m = tile_m + lhi*8 + v; col n = tile_n + lrow
#pragma unroll
  for (int i = 0; i < 2; ++i) {
#pragma unroll
    for (int j = 0; j < 4; ++j) {
      int n = n0 + j * 16 + lrow;
      float bv = bias ? bias[n] : 0.0f;
      float* Crow = C + (size_t)(m0 + i * 16 + lhi * 8) * ldc + n;
#pragma unroll
      for (int v = 0; v < 8; ++v)
        Crow[(size_t)v * ldc] = acc[i][j][v] + bv;
    }
  }
}

// ---------------- attention scores: scores[b,s] = Va_b + sum_h tanh(q[b,h]+U[b,s,h])*Va[h]
__global__ __launch_bounds__(128)
void k_scores(const float* __restrict__ q, const float* __restrict__ U,
              const float* __restrict__ Va_w, const float* __restrict__ Va_b,
              float* __restrict__ scores) {
  int s = blockIdx.x, b = blockIdx.y;
  const float* Urow = U + ((size_t)b * Sc + s) * Hc;
  const float* qrow = q + (size_t)b * Hc;
  float p = 0.f;
  for (int h = threadIdx.x; h < Hc; h += 128)
    p += tanhf(qrow[h] + Urow[h]) * Va_w[h];
  __shared__ float sh[128];
  float r = block_reduce_sum(p, sh);
  if (threadIdx.x == 0) scores[(size_t)b * Sc + s] = r + Va_b[0];
}

// ---------------- softmax over S + context; write attn to d_out, ctx(bf16) into xcat[:,H:2H]
__global__ __launch_bounds__(256)
void k_softmax_ctx(const float* __restrict__ scores, const float* __restrict__ enc,
                   unsigned short* __restrict__ xcat, float* __restrict__ attn_out,
                   int t) {
  int b = blockIdx.x;
  __shared__ float sa[Sc];
  __shared__ float red[256];
  const float* sc = scores + (size_t)b * Sc;
  float v = (threadIdx.x < Sc) ? sc[threadIdx.x] : -1e30f;
  red[threadIdx.x] = v; __syncthreads();
  for (int off = 128; off > 0; off >>= 1) {
    if (threadIdx.x < off) red[threadIdx.x] = fmaxf(red[threadIdx.x], red[threadIdx.x + off]);
    __syncthreads();
  }
  float mx = red[0]; __syncthreads();
  float e = (threadIdx.x < Sc) ? __expf(v - mx) : 0.f;
  red[threadIdx.x] = e; __syncthreads();
  for (int off = 128; off > 0; off >>= 1) {
    if (threadIdx.x < off) red[threadIdx.x] += red[threadIdx.x + off];
    __syncthreads();
  }
  float sum = red[0]; __syncthreads();
  float a = e / sum;
  if (threadIdx.x < Sc) {
    sa[threadIdx.x] = a;
    attn_out[(size_t)b * Tc * Sc + (size_t)t * Sc + threadIdx.x] = a;
  }
  __syncthreads();
  const float* eb = enc + (size_t)b * Sc * Hc;
  for (int h = threadIdx.x; h < Hc; h += 256) {
    float acc = 0.f;
#pragma unroll 4
    for (int s = 0; s < Sc; ++s) acc += sa[s] * eb[(size_t)s * Hc + h];
    xcat[(size_t)b * 3 * Hc + Hc + h] = f32_to_bf16(acc);
  }
}

// ---------------- xcat[:,0:H] = emb[tok], xcat[:,2H:3H] = h_bf ----------------
__global__ __launch_bounds__(256)
void k_embed_h(const unsigned short* __restrict__ emb_bf, const int* __restrict__ target,
               const unsigned short* __restrict__ h_bf, unsigned short* __restrict__ xcat,
               int t) {
  int b = blockIdx.x;
  int tok = (t == 0) ? SOS : target[(size_t)b * Tc + (t - 1)];
  const unsigned short* er = emb_bf + (size_t)tok * Hc;
  unsigned short* xr = xcat + (size_t)b * 3 * Hc;
  for (int h = threadIdx.x; h < Hc; h += 256) {
    xr[h] = er[h];
    xr[2 * Hc + h] = h_bf[(size_t)b * Hc + h];
  }
}

// ---------------- LSTM cell + 3 LayerNorms (fused) ----------------
__global__ __launch_bounds__(256)
void k_lstm_ln(const float* __restrict__ gates, float* __restrict__ c,
               unsigned short* __restrict__ h_bf, unsigned short* __restrict__ out_bf,
               const float* __restrict__ ln_w, const float* __restrict__ ln_b,
               const float* __restrict__ lnh_w, const float* __restrict__ lnh_b,
               float* __restrict__ hT_out, float* __restrict__ cT_out) {
  int b = blockIdx.x;
  const float* g = gates + (size_t)b * 4 * Hc;
  float* cb = c + (size_t)b * Hc;
  float h2v[4], c2v[4];
  float sh = 0.f, sh2 = 0.f, scs = 0.f, sc2 = 0.f;
#pragma unroll
  for (int idx = 0; idx < 4; ++idx) {
    int h = threadIdx.x + idx * 256;
    float ig = sigmoidf_(g[h]);
    float fg = sigmoidf_(g[Hc + h]);
    float gg = tanhf(g[2 * Hc + h]);
    float og = sigmoidf_(g[3 * Hc + h]);
    float c2 = fg * cb[h] + ig * gg;
    float h2 = og * tanhf(c2);
    h2v[idx] = h2; c2v[idx] = c2;
    sh += h2; sh2 += h2 * h2; scs += c2; sc2 += c2 * c2;
  }
  __shared__ float red[256];
  float Sh  = block_reduce_sum(sh,  red);
  float Sh2 = block_reduce_sum(sh2, red);
  float Sc_ = block_reduce_sum(scs, red);
  float Sc2 = block_reduce_sum(sc2, red);
  float mh = Sh / Hc,  vh = Sh2 / Hc - mh * mh;
  float mc = Sc_ / Hc, vc = Sc2 / Hc - mc * mc;
  float rh = rsqrtf(vh + 1e-5f), rc = rsqrtf(vc + 1e-5f);
#pragma unroll
  for (int idx = 0; idx < 4; ++idx) {
    int h = threadIdx.x + idx * 256;
    float nh = (h2v[idx] - mh) * rh;
    float nc = (c2v[idx] - mc) * rc;
    float outv = nh * ln_w[h] + ln_b[h];
    float h3 = nh * lnh_w[h] + lnh_b[h];
    float c3 = nc * lnh_w[h] + lnh_b[h];
    out_bf[(size_t)b * Hc + h] = f32_to_bf16(outv);
    h_bf[(size_t)b * Hc + h]  = f32_to_bf16(h3);
    cb[h] = c3;
    if (hT_out) hT_out[(size_t)b * Hc + h] = h3;
    if (cT_out) cT_out[(size_t)b * Hc + h] = c3;
  }
}

// ---------------- host orchestration ----------------
extern "C" void kernel_launch(void* const* d_in, const int* in_sizes, int n_in,
                              void* d_out, int out_size, void* d_ws, size_t ws_size,
                              hipStream_t stream) {
  const float* enc   = (const float*)d_in[0];
  const float* h0    = (const float*)d_in[1];
  const float* c0    = (const float*)d_in[2];
  const float* emb   = (const float*)d_in[3];
  const float* Wa_w  = (const float*)d_in[4];
  const float* Wa_b  = (const float*)d_in[5];
  const float* Ua_w  = (const float*)d_in[6];
  const float* Ua_b  = (const float*)d_in[7];
  const float* Va_w  = (const float*)d_in[8];
  const float* Va_b  = (const float*)d_in[9];
  const float* w_ih  = (const float*)d_in[10];
  const float* w_hh  = (const float*)d_in[11];
  const float* b_ih  = (const float*)d_in[12];
  const float* b_hh  = (const float*)d_in[13];
  const float* ln_w  = (const float*)d_in[14];
  const float* ln_b  = (const float*)d_in[15];
  const float* lnh_w = (const float*)d_in[16];
  const float* lnh_b = (const float*)d_in[17];
  const float* lin_w = (const float*)d_in[18];
  const float* lin_b = (const float*)d_in[19];
  const int*   target = (const int*)d_in[20];
  float* out = (float*)d_out;

  // workspace carve (256B aligned)
  char* p = (char*)d_ws;
  auto alloc = [&](size_t bytes) -> void* {
    void* r = (void*)p; p += (bytes + 255) & ~(size_t)255; return r;
  };
  unsigned short* enc_bf  = (unsigned short*)alloc((size_t)Bc * Sc * Hc * 2);
  float*          U       = (float*)alloc((size_t)Bc * Sc * Hc * 4);
  unsigned short* Ua_bf   = (unsigned short*)alloc((size_t)Hc * Hc * 2);
  unsigned short* Wa_bf   = (unsigned short*)alloc((size_t)Hc * Hc * 2);
  unsigned short* wcat    = (unsigned short*)alloc((size_t)4 * Hc * 3 * Hc * 2);
  unsigned short* lin_bf  = (unsigned short*)alloc((size_t)Vc * Hc * 2);
  unsigned short* emb_bf  = (unsigned short*)alloc((size_t)Vc * Hc * 2);
  float*          biascat = (float*)alloc((size_t)4 * Hc * 4);
  unsigned short* h_bf    = (unsigned short*)alloc((size_t)Bc * Hc * 2);
  float*          c_ws    = (float*)alloc((size_t)Bc * Hc * 4);
  float*          q_ws    = (float*)alloc((size_t)Bc * Hc * 4);
  float*          scores  = (float*)alloc((size_t)Bc * Sc * 4);
  unsigned short* xcat    = (unsigned short*)alloc((size_t)Bc * 3 * Hc * 2);
  float*          gates   = (float*)alloc((size_t)Bc * 4 * Hc * 4);
  unsigned short* out_bf  = (unsigned short*)alloc((size_t)Bc * Hc * 2);
  (void)ws_size; (void)n_in; (void)in_sizes; (void)out_size;

  auto cvt = [&](const float* s, unsigned short* d, size_t n) {
    k_f32_to_bf16<<<dim3((unsigned)((n + 255) / 256)), 256, 0, stream>>>(s, d, (int)n);
  };

  // ---- one-time prep ----
  cvt(enc,   enc_bf, (size_t)Bc * Sc * Hc);
  cvt(Ua_w,  Ua_bf,  (size_t)Hc * Hc);
  cvt(Wa_w,  Wa_bf,  (size_t)Hc * Hc);
  cvt(lin_w, lin_bf, (size_t)Vc * Hc);
  cvt(emb,   emb_bf, (size_t)Vc * Hc);
  cvt(h0,    h_bf,   (size_t)Bc * Hc);
  hipMemcpyAsync(c_ws, c0, (size_t)Bc * Hc * 4, hipMemcpyDeviceToDevice, stream);
  {
    size_t nw = (size_t)4 * Hc * 3 * Hc;
    k_build_wcat<<<dim3((unsigned)((nw + 255) / 256)), 256, 0, stream>>>(w_ih, w_hh, wcat);
    k_bias_cat<<<dim3(16), 256, 0, stream>>>(b_ih, b_hh, biascat);
  }
  // U = enc_bf @ Ua^T + Ua_b : M=B*S, N=H, K=H  (hoisted out of the time loop)
  k_gemm_bf16<<<dim3(Hc / 128, (Bc * Sc) / 128), 256, 0, stream>>>(
      enc_bf, Hc, Ua_bf, Hc, Ua_b, U, Hc, Hc);

  // ---- time loop ----
  for (int t = 0; t < Tc; ++t) {
    // q = h @ Wa^T + Wa_b : M=B, N=H, K=H
    k_gemm_bf16<<<dim3(Hc / 128, Bc / 128), 256, 0, stream>>>(
        h_bf, Hc, Wa_bf, Hc, Wa_b, q_ws, Hc, Hc);
    // scores[b,s]
    k_scores<<<dim3(Sc, Bc), 128, 0, stream>>>(q_ws, U, Va_w, Va_b, scores);
    // softmax + ctx -> xcat[:,H:2H], attn -> d_out
    k_softmax_ctx<<<dim3(Bc), 256, 0, stream>>>(scores, enc, xcat, out + OUT_ATTN, t);
    // xcat[:,0:H]=emb[tok], xcat[:,2H:3H]=h
    k_embed_h<<<dim3(Bc), 256, 0, stream>>>(emb_bf, target, h_bf, xcat, t);
    // gates = xcat @ wcat^T + (b_ih+b_hh) : M=B, N=4H, K=3H
    k_gemm_bf16<<<dim3(4 * Hc / 128, Bc / 128), 256, 0, stream>>>(
        xcat, 3 * Hc, wcat, 3 * Hc, biascat, gates, 4 * Hc, 3 * Hc);
    // LSTM + layernorms
    bool last = (t == Tc - 1);
    k_lstm_ln<<<dim3(Bc), 256, 0, stream>>>(
        gates, c_ws, h_bf, out_bf, ln_w, ln_b, lnh_w, lnh_b,
        last ? (out + OUT_HT) : nullptr, last ? (out + OUT_CT) : nullptr);
    // logits[:, t, :] = out_ln @ lin_w^T + lin_b : M=B, N=V, K=H, ldc=T*V
    k_gemm_bf16<<<dim3(Vc / 128, Bc / 128), 256, 0, stream>>>(
        out_bf, Hc, lin_bf, Hc, lin_b, out + OUT_LOGITS + (size_t)t * Vc,
        Tc * Vc, Hc);
  }
}
